// CharRNN_1468878815404
// MI455X (gfx1250) — matrix operand order repaired and checked by reading, not compiled, not run on hardware
//
#include <hip/hip_runtime.h>
#include <math.h>

#define SEQ    256
#define BATCH  64
#define EMBED  256
#define HIDDEN 1024
#define GATES  (3 * HIDDEN)
#define NCLS   128

typedef __bf16 bf16_t;
typedef bf16_t          v16bf __attribute__((ext_vector_type(16)));
typedef float           v8f   __attribute__((ext_vector_type(8)));
typedef unsigned short  u16x8 __attribute__((ext_vector_type(8)));

union Frag { v16bf bf; u16x8 h[2]; };

// ---------------- device-global scratch (module-load allocated BSS) ----------
__device__ __attribute__((aligned(16))) unsigned short g_wih0[GATES * EMBED];
__device__ __attribute__((aligned(16))) unsigned short g_whh0[GATES * HIDDEN];
__device__ __attribute__((aligned(16))) unsigned short g_wih1[GATES * HIDDEN];
__device__ __attribute__((aligned(16))) unsigned short g_whh1[GATES * HIDDEN];
__device__ __attribute__((aligned(16))) unsigned short g_wproj[NCLS * HIDDEN];
__device__ __attribute__((aligned(16))) unsigned short g_x[SEQ * BATCH * EMBED];      // embedded input, bf16, [s][b][e]
__device__ __attribute__((aligned(16))) float          g_gi[SEQ * BATCH * GATES];     // precomputed input gates, f32, [s*B+b][g]
__device__ __attribute__((aligned(16))) unsigned short g_out0[SEQ * BATCH * HIDDEN];  // layer-0 outputs bf16
__device__ __attribute__((aligned(16))) unsigned short g_out1[SEQ * BATCH * HIDDEN];  // layer-1 outputs bf16
__device__ __attribute__((aligned(16))) unsigned short g_hbf[2][BATCH * HIDDEN];      // double-buffered bf16 h
__device__ __attribute__((aligned(16))) float          g_h32[2][BATCH * HIDDEN];      // double-buffered f32 h

// ---------------- helpers ----------------------------------------------------
__device__ __forceinline__ unsigned short f2bf(float f) {
  unsigned u = __builtin_bit_cast(unsigned, f);
  u += 0x7FFFu + ((u >> 16) & 1u);           // round-to-nearest-even
  return (unsigned short)(u >> 16);
}

__device__ __forceinline__ float sigmoidf(float x) {
  return 1.0f / (1.0f + __expf(-x));
}

// A fragment, 16x32 bf16 (M x K), row-major src with leading dim ld.
// ISA layout: lanes 0-15 m=lane, VGPR0-3 K=0..7, VGPR4-7 K=16..23;
//             lanes 16-31 same m, K=8..15 / K=24..31.
__device__ __forceinline__ v16bf load_a(const unsigned short* __restrict__ base,
                                        int ld, int rowBase, int kBase, int lane) {
  const int m    = lane & 15;
  const int half = lane >> 4;
  const unsigned short* p = base + (size_t)(rowBase + m) * ld + kBase + half * 8;
  Frag f;
  f.h[0] = *(const u16x8*)(p);       // K = kBase + 8*half + 0..7
  f.h[1] = *(const u16x8*)(p + 16);  // K = kBase + 16 + 8*half + 0..7
  return f.bf;
}

// B fragment, 32x16 bf16 (K x N). Column n held by lane n; lanes 0-15 K=kBase+0..7,16..23,
// lanes 16-31 K=kBase+8..15,24..31 (same order as A).  B[k][n] = W[colBase+n][k].
__device__ __forceinline__ v16bf load_b(const unsigned short* __restrict__ W,
                                        int ld, int colBase, int kBase, int lane) {
  const int n    = lane & 15;
  const int half = lane >> 4;
  const unsigned short* p = W + (size_t)(colBase + n) * ld + kBase + half * 8;
  Frag f;
  f.h[0] = *(const u16x8*)(p);
  f.h[1] = *(const u16x8*)(p + 16);
  return f.bf;
}

__device__ __forceinline__ v8f wmma_bf16(v16bf a, v16bf b, v8f c) {
  return __builtin_amdgcn_wmma_f32_16x16x32_bf16(false, a, false, b, (short)0, c,
                                                 false, false);
}

// ---------------- kernels ----------------------------------------------------

// f32 -> bf16 weight conversion into a selected device-global array
__global__ void k_cvt(const float* __restrict__ src, int n, int which) {
  unsigned short* dst;
  switch (which) {
    case 0: dst = g_wih0; break;
    case 1: dst = g_whh0; break;
    case 2: dst = g_wih1; break;
    case 3: dst = g_whh1; break;
    default: dst = g_wproj; break;
  }
  int i = blockIdx.x * blockDim.x + threadIdx.x;
  if (i < n) dst[i] = f2bf(src[i]);
}

// embedding gather: x[B][S] -> g_x[s][b][e] (bf16, time-major)
__global__ void k_embed(const int* __restrict__ x, const float* __restrict__ embed) {
  int i = blockIdx.x * blockDim.x + threadIdx.x;
  if (i >= SEQ * BATCH * EMBED) return;
  int e  = i % EMBED;
  int sb = i / EMBED;           // s*BATCH + b
  int b  = sb % BATCH;
  int s  = sb / BATCH;
  int cls = x[b * SEQ + s];
  g_x[i] = f2bf(embed[(size_t)cls * EMBED + e]);
}

// gi = X @ W_ih^T + b_ih   (all timesteps at once; M = S*B = 16384, N = 3072)
// one wave computes a 16(M) x 64(N) tile: 4 accumulators share one A fragment.
__global__ void __launch_bounds__(256)
k_gemm_gi(int K, int srcSel, int wSel, const float* __restrict__ bias) {
  const unsigned short* X = (srcSel == 0) ? g_x    : g_out0;
  const unsigned short* W = (wSel  == 0) ? g_wih0 : g_wih1;
  const int lane   = threadIdx.x & 31;
  const int waveId = blockIdx.x * 8 + (threadIdx.x >> 5);
  const int nGroups = GATES / 64;                 // 48
  const int mTile  = waveId / nGroups;
  const int nGroup = waveId % nGroups;
  if (mTile >= (SEQ * BATCH) / 16) return;        // wave-uniform
  const int rowBase = mTile * 16;
  const int gBase   = nGroup * 64;

  v8f acc[4] = {v8f{}, v8f{}, v8f{}, v8f{}};
  for (int kb = 0; kb < K; kb += 32) {
    v16bf a  = load_a(X, K, rowBase, kb, lane);
    v16bf b0 = load_b(W, K, gBase +  0, kb, lane);
    v16bf b1 = load_b(W, K, gBase + 16, kb, lane);
    v16bf b2 = load_b(W, K, gBase + 32, kb, lane);
    v16bf b3 = load_b(W, K, gBase + 48, kb, lane);
    acc[0] = wmma_bf16(a, b0, acc[0]);
    acc[1] = wmma_bf16(a, b1, acc[1]);
    acc[2] = wmma_bf16(a, b2, acc[2]);
    acc[3] = wmma_bf16(a, b3, acc[3]);
  }
  const int n = lane & 15, half = lane >> 4;
#pragma unroll
  for (int c = 0; c < 4; ++c) {
    const int g = gBase + c * 16 + n;
    const float bb = bias[g];
#pragma unroll
    for (int i = 0; i < 8; ++i) {
      const int r = rowBase + i + 8 * half;
      g_gi[(size_t)r * GATES + g] = acc[c][i] + bb;
    }
  }
}

// init h buffers from hs[layer]
__global__ void k_init_h(const float* __restrict__ hs, int layer) {
  int i = blockIdx.x * blockDim.x + threadIdx.x;
  if (i >= BATCH * HIDDEN) return;
  float v = hs[(size_t)layer * BATCH * HIDDEN + i];
  g_h32[0][i] = v;
  g_hbf[0][i] = f2bf(v);
}

// one recurrent step, fully fused: gh = h @ W_hh^T (WMMA, 3 gate accumulators
// sharing one A fragment), then gate nonlinearities + h update from registers.
// 256 waves: mTile in [0,4) x nTile in [0,64).
__global__ void __launch_bounds__(256)
k_gru_step(const float* __restrict__ b_hh, int layer, int t) {
  const unsigned short* Whh = (layer == 0) ? g_whh0 : g_whh1;
  unsigned short* outBuf    = (layer == 0) ? g_out0 : g_out1;
  const unsigned short* hin = g_hbf[t & 1];
  unsigned short*      hout = g_hbf[(t + 1) & 1];
  const float*        h32in = g_h32[t & 1];
  float*             h32out = g_h32[(t + 1) & 1];

  const int lane   = threadIdx.x & 31;
  const int waveId = blockIdx.x * 8 + (threadIdx.x >> 5);
  const int mTile  = waveId >> 6;   // 0..3   (batch tiles)
  const int nTile  = waveId & 63;   // 0..63  (hidden tiles)
  const int rowBase = mTile * 16;
  const int jBase   = nTile * 16;

  v8f aR = v8f{}, aZ = v8f{}, aN = v8f{};
  for (int kb = 0; kb < HIDDEN; kb += 32) {
    v16bf a  = load_a(hin, HIDDEN, rowBase, kb, lane);
    v16bf br = load_b(Whh, HIDDEN,              jBase, kb, lane);
    v16bf bz = load_b(Whh, HIDDEN,     HIDDEN + jBase, kb, lane);
    v16bf bn = load_b(Whh, HIDDEN, 2 * HIDDEN + jBase, kb, lane);
    aR = wmma_bf16(a, br, aR);
    aZ = wmma_bf16(a, bz, aZ);
    aN = wmma_bf16(a, bn, aN);
  }

  const int n = lane & 15, half = lane >> 4;
  const int j = jBase + n;
  const float bhr = b_hh[j];
  const float bhz = b_hh[HIDDEN + j];
  const float bhn = b_hh[2 * HIDDEN + j];
  const float* gi_t = g_gi + (size_t)t * BATCH * GATES;

#pragma unroll
  for (int i = 0; i < 8; ++i) {
    const int b = rowBase + i + 8 * half;
    const float* gi_b = gi_t + (size_t)b * GATES;
    const float r  = sigmoidf(gi_b[j]              + aR[i] + bhr);
    const float z  = sigmoidf(gi_b[HIDDEN + j]     + aZ[i] + bhz);
    const float nn = tanhf   (gi_b[2 * HIDDEN + j] + r * (aN[i] + bhn));
    const float hprev = h32in[(size_t)b * HIDDEN + j];
    const float hn = (1.0f - z) * nn + z * hprev;
    h32out[(size_t)b * HIDDEN + j] = hn;
    const unsigned short hb = f2bf(hn);
    hout[(size_t)b * HIDDEN + j] = hb;
    outBuf[((size_t)t * BATCH + b) * HIDDEN + j] = hb;
  }
}

// copy final f32 hidden state (lives in buffer 0 after an even number of steps)
__global__ void k_copy_h(float* __restrict__ dst) {
  int i = blockIdx.x * blockDim.x + threadIdx.x;
  if (i < BATCH * HIDDEN) dst[i] = g_h32[0][i];
}

// logits = out1 @ w_proj^T + b_proj, remapped from [s][b] rows to [b*S+s][c]
__global__ void __launch_bounds__(256)
k_proj(const float* __restrict__ b_proj, float* __restrict__ logits) {
  const int lane   = threadIdx.x & 31;
  const int waveId = blockIdx.x * 8 + (threadIdx.x >> 5);
  const int nGroups = NCLS / 64;                  // 2
  const int mTile  = waveId / nGroups;
  const int nGroup = waveId % nGroups;
  if (mTile >= (SEQ * BATCH) / 16) return;
  const int rowBase = mTile * 16;
  const int cBase   = nGroup * 64;

  v8f acc[4] = {v8f{}, v8f{}, v8f{}, v8f{}};
  for (int kb = 0; kb < HIDDEN; kb += 32) {
    v16bf a = load_a(g_out1, HIDDEN, rowBase, kb, lane);
#pragma unroll
    for (int c = 0; c < 4; ++c) {
      v16bf b = load_b(g_wproj, HIDDEN, cBase + c * 16, kb, lane);
      acc[c] = wmma_bf16(a, b, acc[c]);
    }
  }
  const int n = lane & 15, half = lane >> 4;
#pragma unroll
  for (int c = 0; c < 4; ++c) {
    const int cc = cBase + c * 16 + n;
    const float bb = b_proj[cc];
#pragma unroll
    for (int i = 0; i < 8; ++i) {
      const int r = rowBase + i + 8 * half;  // r = s*BATCH + b
      const int s = r >> 6;
      const int b = r & 63;
      logits[((size_t)b * SEQ + s) * NCLS + cc] = acc[c][i] + bb;
    }
  }
}

// ---------------- host launcher ----------------------------------------------
extern "C" void kernel_launch(void* const* d_in, const int* in_sizes, int n_in,
                              void* d_out, int out_size, void* d_ws, size_t ws_size,
                              hipStream_t stream) {
  (void)in_sizes; (void)n_in; (void)d_ws; (void)ws_size; (void)out_size;

  const int*   x      = (const int*)  d_in[0];
  const float* hs     = (const float*)d_in[1];
  const float* embed  = (const float*)d_in[2];
  const float* w_ih0  = (const float*)d_in[3];
  const float* w_hh0  = (const float*)d_in[4];
  const float* b_ih0  = (const float*)d_in[5];
  const float* b_hh0  = (const float*)d_in[6];
  const float* w_ih1  = (const float*)d_in[7];
  const float* w_hh1  = (const float*)d_in[8];
  const float* b_ih1  = (const float*)d_in[9];
  const float* b_hh1  = (const float*)d_in[10];
  const float* w_proj = (const float*)d_in[11];
  const float* b_proj = (const float*)d_in[12];

  float* logits = (float*)d_out;
  float* hT     = logits + (size_t)BATCH * SEQ * NCLS;

  const int T = 256;
  auto blocks = [](int n) { return (n + 255) / 256; };

  // 1) weights -> bf16
  k_cvt<<<blocks(GATES * EMBED),  T, 0, stream>>>(w_ih0,  GATES * EMBED,  0);
  k_cvt<<<blocks(GATES * HIDDEN), T, 0, stream>>>(w_hh0,  GATES * HIDDEN, 1);
  k_cvt<<<blocks(GATES * HIDDEN), T, 0, stream>>>(w_ih1,  GATES * HIDDEN, 2);
  k_cvt<<<blocks(GATES * HIDDEN), T, 0, stream>>>(w_hh1,  GATES * HIDDEN, 3);
  k_cvt<<<blocks(NCLS * HIDDEN),  T, 0, stream>>>(w_proj, NCLS * HIDDEN,  4);

  // 2) embedding gather (time-major bf16)
  k_embed<<<blocks(SEQ * BATCH * EMBED), T, 0, stream>>>(x, embed);

  const int giBlocks   = ((SEQ * BATCH) / 16) * (GATES / 64) / 8;  // 6144
  const int stepBlocks = (4 * 64) / 8;                             // 32
  const int projBlocks = ((SEQ * BATCH) / 16) * (NCLS / 64) / 8;   // 256

  // ---- layer 0 ----
  k_gemm_gi<<<giBlocks, T, 0, stream>>>(EMBED, /*src=*/0, /*w=*/0, b_ih0);
  k_init_h<<<blocks(BATCH * HIDDEN), T, 0, stream>>>(hs, 0);
  for (int t = 0; t < SEQ; ++t)
    k_gru_step<<<stepBlocks, T, 0, stream>>>(b_hh0, /*layer=*/0, t);
  k_copy_h<<<blocks(BATCH * HIDDEN), T, 0, stream>>>(hT);                       // hT[0]

  // ---- layer 1 ----
  k_gemm_gi<<<giBlocks, T, 0, stream>>>(HIDDEN, /*src=*/1, /*w=*/1, b_ih1);
  k_init_h<<<blocks(BATCH * HIDDEN), T, 0, stream>>>(hs, 1);
  for (int t = 0; t < SEQ; ++t)
    k_gru_step<<<stepBlocks, T, 0, stream>>>(b_hh1, /*layer=*/1, t);
  k_copy_h<<<blocks(BATCH * HIDDEN), T, 0, stream>>>(hT + BATCH * HIDDEN);      // hT[1]

  // ---- projection ----
  k_proj<<<projBlocks, T, 0, stream>>>(b_proj, logits);
}
